// SpGAT_31430570672189
// MI455X (gfx1250) — compile-verified, matrix operand order checked by reading, not run on hardware
//
#include <hip/hip_runtime.h>
#include <hip/hip_bf16.h>

// ---------------------------------------------------------------------------
// KBGAT-style sparse graph attention for MI455X (gfx1250, wave32, WMMA).
// Per-edge projections run on v_wmma_f32_16x16x32_f16 (f16 in, f32 acc);
// edge->node aggregation uses global f32 atomics (L2-resident accumulators).
// Gathers are vectorized: one coalesced b128 wave-load per 512B feature row.
// ---------------------------------------------------------------------------

typedef __attribute__((ext_vector_type(16))) _Float16 v16h;
typedef __attribute__((ext_vector_type(8)))  _Float16 v8h;
typedef __attribute__((ext_vector_type(4)))  _Float16 v4h;
typedef __attribute__((ext_vector_type(8)))  float    v8f;

#define N_NODES 50000
#define N_REL   474
#define NFEAT   128
#define E_1HOP  100000
#define E_NHOP  20000
#define E_TOT   (E_1HOP + E_NHOP)   // 120000 edges per direction
#define KDIM    384                 // 2*NFEAT + REL_DIM
#define KSTEPS  12                  // 384 / 32
#define ALPHA_LRELU 0.2f

__device__ __forceinline__ float eluf(float v) {
    return (v > 0.f) ? v : (__expf(v) - 1.f);
}

// ------------------------- prep: concatenated edge arrays -------------------
__global__ __launch_bounds__(256) void prep_edges_kernel(
    const int* __restrict__ edge_list,            // [2, E_1HOP]
    const int* __restrict__ edge_list_nhop,       // [2, E_NHOP]
    const int* __restrict__ edge_list_rev,        // [2, E_1HOP]
    const int* __restrict__ edge_list_rev_nhop,   // [2, E_NHOP]
    int* __restrict__ srcf, int* __restrict__ dstf,
    int* __restrict__ srcr, int* __restrict__ dstr)
{
    int e = blockIdx.x * blockDim.x + threadIdx.x;
    if (e >= E_TOT) return;
    if (e < E_1HOP) {
        srcf[e] = edge_list[e];
        dstf[e] = edge_list[E_1HOP + e];
        srcr[e] = edge_list_rev[e];
        dstr[e] = edge_list_rev[E_1HOP + e];
    } else {
        int i = e - E_1HOP;
        srcf[e] = edge_list_nhop[i];
        dstf[e] = edge_list_nhop[E_NHOP + i];
        srcr[e] = edge_list_rev_nhop[i];
        dstr[e] = edge_list_rev_nhop[E_NHOP + i];
    }
}

// --------------------- prep: per-edge relation embeddings (f16) -------------
// rel_src: [nrel, 128] f32 source table; writes eF/eR [E_TOT, 128] f16.
__global__ __launch_bounds__(256) void prep_eemb_kernel(
    const float* __restrict__ rel_src,
    const int* __restrict__ edge_type,           // [E_1HOP]
    const int* __restrict__ edge_type_nhop,      // [E_NHOP, 2]
    const int* __restrict__ edge_type_rev_nhop,  // [E_NHOP, 2]
    const float* __restrict__ w1, const float* __restrict__ w2,
    _Float16* __restrict__ eF, _Float16* __restrict__ eR)
{
    long idx = (long)blockIdx.x * blockDim.x + threadIdx.x;
    if (idx >= (long)E_TOT * 128) return;
    int e = (int)(idx >> 7);
    int k = (int)(idx & 127);
    float vf, vr;
    if (e < E_1HOP) {
        int t = edge_type[e];
        vf = vr = rel_src[(long)t * 128 + k];
    } else {
        int i = e - E_1HOP;
        int t0 = edge_type_nhop[i * 2 + 0];
        int t1 = edge_type_nhop[i * 2 + 1];
        vf = rel_src[(long)t0 * 128 + k] * w1[k] + rel_src[(long)t1 * 128 + k] * w2[k];
        int r0 = edge_type_rev_nhop[i * 2 + 0];
        int r1 = edge_type_rev_nhop[i * 2 + 1];
        vr = rel_src[(long)r0 * 128 + k] * w1[k] + rel_src[(long)r1 * 128 + k] * w2[k];
    }
    eF[idx] = (_Float16)vf;
    eR[idx] = (_Float16)vr;
}

// -------------------- prep: out_relation_1 = relation_embed @ W -------------
__global__ __launch_bounds__(256) void relproj_kernel(
    const float* __restrict__ relemb,  // [474, 128]
    const float* __restrict__ W,       // [128, 128]
    float* __restrict__ orel)          // [474, 128]
{
    int idx = blockIdx.x * blockDim.x + threadIdx.x;
    if (idx >= N_REL * 128) return;
    int r = idx >> 7, c = idx & 127;
    float s = 0.f;
    #pragma unroll 8
    for (int k = 0; k < 128; ++k)
        s += relemb[r * 128 + k] * W[k * 128 + c];
    orel[idx] = s;
}

// -------- prep: pack a^T into the wmma_f32_16x16x32_f16 B-fragment layout ---
// B is KxN; B[k][n] = a[n][k].  Per lane l: N = l&15; lane<16 holds K 0..7 &
// 16..23 of the 32-slice, lane>=16 holds K 8..15 & 24..31 (16 halves / lane).
__global__ __launch_bounds__(256) void pack_b_kernel(
    const float* __restrict__ amat,    // [ntiles*16, 384]
    _Float16* __restrict__ bpack,      // [ntiles*KSTEPS*32*16]
    int ntiles)
{
    int idx = blockIdx.x * blockDim.x + threadIdx.x;
    int total = ntiles * KSTEPS * 32 * 16;
    if (idx >= total) return;
    int j    = idx & 15;
    int l    = (idx >> 4) & 31;
    int rest = idx >> 9;
    int ks   = rest % KSTEPS;
    int nt   = rest / KSTEPS;
    int n = nt * 16 + (l & 15);
    int k = ks * 32 + (j & 7) + ((j >> 3) * 16) + ((l >> 4) * 8);
    bpack[idx] = (_Float16)amat[(long)n * KDIM + k];
}

// --------------------------- attend: the hot kernel -------------------------
// One wave per 16-edge tile.  em = [x_src;x_dst;rel] @ a^T via WMMA, then
// w = exp(-leakyrelu(em . a2)), scatter w*em / w with global f32 atomics.
template <int NT>
__global__ __launch_bounds__(32) void attend_kernel(
    const float*    __restrict__ x,     // [N_NODES, 128]
    const _Float16* __restrict__ eemb,  // [E_TOT, 128]
    const int*      __restrict__ src,
    const int*      __restrict__ dst,
    const _Float16* __restrict__ bpack, // packed B fragments
    const float*    __restrict__ a2,    // [NT*16]
    float* __restrict__ acc,            // [N_NODES, NT*16]
    float* __restrict__ den,            // [N_NODES]
    int n_edges)
{
    constexpr int NCOL = NT * 16;
    __shared__ __align__(32) _Float16 Alds[16 * KDIM];
    __shared__ float em_lds[16 * NCOL];
    __shared__ float w_lds[16];
    __shared__ int   node_lds[16];

    const int lane = threadIdx.x;
    const int e0 = blockIdx.x * 16;

    // ---- gather A tile: [x_src ; x_dst ; rel] -> LDS f16 -------------------
    // Each 512B feature row is one coalesced wave load (32 lanes x float4);
    // the f16 relation row is one 256B wave load (32 lanes x v4h).
    #pragma unroll 4
    for (int el = 0; el < 16; ++el) {
        int e = e0 + el;
        v4h hs = {}, hd = {}, hr = {};
        if (e < n_edges) {
            const float4 vs = *(const float4*)(x + (long)src[e] * 128 + lane * 4);
            const float4 vd = *(const float4*)(x + (long)dst[e] * 128 + lane * 4);
            hr = *(const v4h*)(eemb + (long)e * 128 + lane * 4);
            hs[0] = (_Float16)vs.x; hs[1] = (_Float16)vs.y;
            hs[2] = (_Float16)vs.z; hs[3] = (_Float16)vs.w;
            hd[0] = (_Float16)vd.x; hd[1] = (_Float16)vd.y;
            hd[2] = (_Float16)vd.z; hd[3] = (_Float16)vd.w;
        }
        *(v4h*)(&Alds[el * KDIM +       lane * 4]) = hs;
        *(v4h*)(&Alds[el * KDIM + 128 + lane * 4]) = hd;
        *(v4h*)(&Alds[el * KDIM + 256 + lane * 4]) = hr;
    }
    if (lane < 16) {
        int e = e0 + lane;
        node_lds[lane] = (e < n_edges) ? src[e] : 0;
    }
    __syncthreads();

    // ---- WMMA: 16(edges) x NCOL, K = 384 in 12 steps of 32 ----
    v8f accf[NT];
    #pragma unroll
    for (int nt = 0; nt < NT; ++nt) { v8f z = {}; accf[nt] = z; }

    const int m   = lane & 15;
    const int klo = (lane >> 4) * 8;
    #pragma unroll
    for (int ks = 0; ks < KSTEPS; ++ks) {
        union { v16h v; v8h h[2]; } af;
        const _Float16* abase = &Alds[m * KDIM + ks * 32 + klo];
        af.h[0] = *(const v8h*)(abase);
        af.h[1] = *(const v8h*)(abase + 16);
        #pragma unroll
        for (int nt = 0; nt < NT; ++nt) {
            v16h bf = *(const v16h*)(bpack + (((long)(nt * KSTEPS + ks) * 32 + lane) * 16));
            accf[nt] = __builtin_amdgcn_wmma_f32_16x16x32_f16(
                false, af.v, false, bf, (short)0, accf[nt], false, false);
        }
    }

    // ---- spill em tile (C/D layout: vgpr j -> row j or j+8, col = lane&15) --
    const int rowbase = (lane >> 4) * 8;
    const int col0 = lane & 15;
    #pragma unroll
    for (int nt = 0; nt < NT; ++nt) {
        #pragma unroll
        for (int j = 0; j < 8; ++j)
            em_lds[(rowbase + j) * NCOL + nt * 16 + col0] = accf[nt][j];
    }
    __syncthreads();

    // ---- per-edge score & softmax weight (global max-shift cancels) --------
    // Lane l handles edge (l&15); the two 16-lane halves split the dot
    // product over columns and combine with a cross-half shuffle.
    {
        const int el   = lane & 15;
        const int half = lane >> 4;
        float s = 0.f;
        #pragma unroll 8
        for (int n = half * (NCOL / 2); n < (half + 1) * (NCOL / 2); ++n)
            s += em_lds[el * NCOL + n] * a2[n];
        s += __shfl_xor(s, 16, 32);
        if (lane < 16) {
            int e = e0 + lane;
            float w = 0.f;
            if (e < n_edges) {
                float p = (s > 0.f) ? -s : -(ALPHA_LRELU * s);
                w = __expf(p);
            }
            w_lds[lane] = w;
        }
    }
    __syncthreads();

    // ---- scatter: segment-sum on edges[0] via f32 atomics (hits L2) --------
    for (int el = 0; el < 16; ++el) {
        int e = e0 + el;
        if (e >= n_edges) break;
        float w = w_lds[el];
        int node = node_lds[el];
        for (int n = lane; n < NCOL; n += 32)
            atomicAdd(&acc[(long)node * NCOL + n], w * em_lds[el * NCOL + n]);
        if (lane == 0)
            atomicAdd(&den[node], w);
    }
}

// --------------------------- finalize layer 1 -------------------------------
// x1[n, h*64+c] = elu(num_f/den_f + num_r/den_r)
__global__ __launch_bounds__(256) void finalize1_kernel(
    const float* __restrict__ acc1,  // [head][dir][N][64]
    const float* __restrict__ den1,  // [head*2+dir][N]
    float* __restrict__ x1)          // [N, 128]
{
    long idx = (long)blockIdx.x * blockDim.x + threadIdx.x;
    if (idx >= (long)N_NODES * 128) return;
    int n = (int)(idx >> 7);
    int c = (int)(idx & 127);
    int h = c >> 6, cc = c & 63;
    long bf = ((long)(h * 2 + 0) * N_NODES + n);
    long br = ((long)(h * 2 + 1) * N_NODES + n);
    float hf = acc1[bf * 64 + cc] / fmaxf(den1[(long)(h * 2 + 0) * N_NODES + n], 1e-12f);
    float hr = acc1[br * 64 + cc] / fmaxf(den1[(long)(h * 2 + 1) * N_NODES + n], 1e-12f);
    x1[idx] = eluf(hf + hr);
}

// --------------------------- finalize layer 2 -------------------------------
__global__ __launch_bounds__(256) void finalize2_kernel(
    const float* __restrict__ acc2,  // [dir][N][128]
    const float* __restrict__ den2,  // [dir][N]
    float* __restrict__ x2)          // [N, 128]
{
    long idx = (long)blockIdx.x * blockDim.x + threadIdx.x;
    if (idx >= (long)N_NODES * 128) return;
    int n = (int)(idx >> 7);
    int c = (int)(idx & 127);
    float hf = acc2[(long)n * 128 + c] / fmaxf(den2[n], 1e-12f);
    float hr = acc2[(long)N_NODES * 128 + (long)n * 128 + c] /
               fmaxf(den2[N_NODES + n], 1e-12f);
    x2[idx] = eluf(hf + hr);
}

// ---------------------------------------------------------------------------
extern "C" void kernel_launch(void* const* d_in, const int* in_sizes, int n_in,
                              void* d_out, int out_size, void* d_ws, size_t ws_size,
                              hipStream_t stream) {
    const float* x_emb   = (const float*)d_in[0];
    const float* relemb  = (const float*)d_in[1];
    const float* W       = (const float*)d_in[2];
    const float* W_1     = (const float*)d_in[3];
    const float* W_2     = (const float*)d_in[4];
    const float* W_1f    = (const float*)d_in[5];
    const float* W_2f    = (const float*)d_in[6];
    const float* a_mat   = (const float*)d_in[7];   // [2, 64, 384]
    const float* a2_vec  = (const float*)d_in[8];   // [2, 64]
    const float* a_out   = (const float*)d_in[9];   // [128, 384]
    const float* a2_out  = (const float*)d_in[10];  // [128]
    const int* edge_list          = (const int*)d_in[11];
    const int* edge_type          = (const int*)d_in[12];
    const int* edge_list_rev      = (const int*)d_in[13];
    const int* edge_list_nhop     = (const int*)d_in[14];
    const int* edge_type_nhop     = (const int*)d_in[15];
    const int* edge_list_rev_nhop = (const int*)d_in[16];
    const int* edge_type_rev_nhop = (const int*)d_in[17];

    float* out_x2 = (float*)d_out;                       // [N, 128]
    float* orel   = out_x2 + (long)N_NODES * 128;        // [474, 128]

    // ---- workspace layout (256B aligned, sequential) ----
    char* ws = (char*)d_ws;
    size_t off = 0;
    auto take = [&](size_t bytes) { char* p = ws + off; off += (bytes + 255) & ~(size_t)255; return p; };
    _Float16* eembF  = (_Float16*)take((size_t)E_TOT * 128 * 2);
    _Float16* eembR  = (_Float16*)take((size_t)E_TOT * 128 * 2);
    _Float16* ee2F   = (_Float16*)take((size_t)E_TOT * 128 * 2);
    _Float16* ee2R   = (_Float16*)take((size_t)E_TOT * 128 * 2);
    float*    x1     = (float*)take((size_t)N_NODES * 128 * 4);
    float*    acc1   = (float*)take((size_t)4 * N_NODES * 64 * 4);  // [h][d][N][64]
    float*    den1   = (float*)take((size_t)4 * N_NODES * 4);
    float*    acc2   = (float*)take((size_t)2 * N_NODES * 128 * 4); // [d][N][128]
    float*    den2   = (float*)take((size_t)2 * N_NODES * 4);
    size_t zero_base = (char*)acc1 - ws;
    size_t zero_len  = ((char*)den2 + (size_t)2 * N_NODES * 4) - (char*)acc1;
    int* srcf = (int*)take((size_t)E_TOT * 4);
    int* dstf = (int*)take((size_t)E_TOT * 4);
    int* srcr = (int*)take((size_t)E_TOT * 4);
    int* dstr = (int*)take((size_t)E_TOT * 4);
    _Float16* bpack1 = (_Float16*)take((size_t)2 * 4 * KSTEPS * 32 * 16 * 2); // 2 heads, NT=4
    _Float16* bpack2 = (_Float16*)take((size_t)8 * KSTEPS * 32 * 16 * 2);     // NT=8
    (void)ws_size; (void)in_sizes; (void)n_in; (void)out_size;

    // 0) zero node accumulators (replayed every graph launch)
    hipMemsetAsync(ws + zero_base, 0, zero_len, stream);

    // 1) concatenated edge arrays
    prep_edges_kernel<<<(E_TOT + 255) / 256, 256, 0, stream>>>(
        edge_list, edge_list_nhop, edge_list_rev, edge_list_rev_nhop,
        srcf, dstf, srcr, dstr);

    // 2) layer-1 per-edge relation embeddings (f16)
    {
        long tot = (long)E_TOT * 128;
        prep_eemb_kernel<<<(unsigned)((tot + 255) / 256), 256, 0, stream>>>(
            relemb, edge_type, edge_type_nhop, edge_type_rev_nhop,
            W_1, W_2, eembF, eembR);
    }

    // 3) out_relation_1 (straight into d_out tail)
    relproj_kernel<<<(N_REL * 128 + 255) / 256, 256, 0, stream>>>(relemb, W, orel);

    // 4) layer-2 per-edge relation embeddings (reads orel; stream-ordered)
    {
        long tot = (long)E_TOT * 128;
        prep_eemb_kernel<<<(unsigned)((tot + 255) / 256), 256, 0, stream>>>(
            orel, edge_type, edge_type_nhop, edge_type_rev_nhop,
            W_1f, W_2f, ee2F, ee2R);
    }

    // 5) prepack B fragments (a^T per head, a_out^T)
    {
        int tot1 = 4 * KSTEPS * 32 * 16;
        pack_b_kernel<<<(tot1 + 255) / 256, 256, 0, stream>>>(a_mat, bpack1, 4);
        pack_b_kernel<<<(tot1 + 255) / 256, 256, 0, stream>>>(
            a_mat + (long)64 * KDIM, bpack1 + tot1, 4);
        int tot2 = 8 * KSTEPS * 32 * 16;
        pack_b_kernel<<<(tot2 + 255) / 256, 256, 0, stream>>>(a_out, bpack2, 8);
    }

    const int ntiles_e = (E_TOT + 15) / 16;  // 7500 waves per attend

    // 6) layer-1 attention: 2 heads x {forward, reverse}
    for (int h = 0; h < 2; ++h) {
        const _Float16* bp = bpack1 + (long)h * 4 * KSTEPS * 32 * 16;
        const float* a2h = a2_vec + h * 64;
        attend_kernel<4><<<ntiles_e, 32, 0, stream>>>(
            x_emb, eembF, srcf, dstf, bp, a2h,
            acc1 + (long)(h * 2 + 0) * N_NODES * 64, den1 + (long)(h * 2 + 0) * N_NODES,
            E_TOT);
        attend_kernel<4><<<ntiles_e, 32, 0, stream>>>(
            x_emb, eembR, srcr, dstr, bp, a2h,
            acc1 + (long)(h * 2 + 1) * N_NODES * 64, den1 + (long)(h * 2 + 1) * N_NODES,
            E_TOT);
    }

    // 7) x1 = elu(hf + hr), heads concatenated
    finalize1_kernel<<<(unsigned)(((long)N_NODES * 128 + 255) / 256), 256, 0, stream>>>(
        acc1, den1, x1);

    // 8) layer-2 attention: {forward, reverse}
    attend_kernel<8><<<ntiles_e, 32, 0, stream>>>(
        x1, ee2F, srcf, dstf, bpack2, a2_out,
        acc2, den2, E_TOT);
    attend_kernel<8><<<ntiles_e, 32, 0, stream>>>(
        x1, ee2R, srcr, dstr, bpack2, a2_out,
        acc2 + (long)N_NODES * 128, den2 + N_NODES, E_TOT);

    // 9) x2 = elu(hf + hr)
    finalize2_kernel<<<(unsigned)(((long)N_NODES * 128 + 255) / 256), 256, 0, stream>>>(
        acc2, den2, out_x2);
}
